// Block_52278341927299
// MI455X (gfx1250) — matrix operand order repaired and checked by reading, not compiled, not run on hardware
//
#include <hip/hip_runtime.h>
#include <hip/hip_bf16.h>
#include <math.h>

typedef __attribute__((ext_vector_type(16))) __bf16 v16bf;
typedef __attribute__((ext_vector_type(8)))  __bf16 v8bf;
typedef __attribute__((ext_vector_type(8)))  float  v8f;
typedef __attribute__((ext_vector_type(4)))  unsigned int v4u;
typedef __attribute__((ext_vector_type(8)))  int v8i;
typedef __attribute__((ext_vector_type(4)))  int v4i;

#define LL    8192
#define CCH   512
#define MCH   1024
#define KDEP  1152

#define EPSN         1.0e-4f
#define INV_SQRT512  0.044194173824159216f
#define INV_SQRT1024 0.03125f
#define INV_SQRT1152 0.029462782549439476f
#define INV_SQRT128  0.08838834764831845f
#define INV_SQRT2    0.7071067811865476f
#define INV_SILU     1.6778523489932886f   // 1/0.596
#define CLIPV        256.0f

// ---------------------------------------------------------------- WMMA helpers

__device__ __forceinline__ v8f wmma_bf16(v16bf a, v16bf b, v8f c) {
  return __builtin_amdgcn_wmma_f32_16x16x32_bf16(false, a, false, b, (short)0, c,
                                                 false, false);
}

// A fragment: 16x32 (MxK). lane<16: row=lane, K[0..7]+K[16..23];
// lane>=16: row=lane-16, K[8..15]+K[24..31]. (ISA 7.12.2, 16-bit A 16x32)
__device__ __forceinline__ v16bf frag_a(const __bf16* Abase, int lda, int lane, int k0) {
  int m  = lane & 15;
  int ko = (lane < 16) ? 0 : 8;
  const __bf16* p = Abase + (size_t)m * lda + k0 + ko;
  v8bf lo = *(const v8bf*)p;
  v8bf hi = *(const v8bf*)(p + 16);
  v16bf a;
#pragma unroll
  for (int i = 0; i < 8; ++i) { a[i] = lo[i]; a[i + 8] = hi[i]; }
  return a;
}

// B fragment: 32x16 (KxN). lane<16: col=lane, K[0..7]+K[16..23]; lane>=16: col=lane-16, K[8..15]+K[24..31].
__device__ __forceinline__ v16bf frag_b(const __bf16* Bbase, int ldb, int lane, int k0) {
  int n  = lane & 15;
  int ko = (lane < 16) ? 0 : 8;
  const __bf16* p = Bbase + (size_t)n * ldb + k0 + ko;
  v8bf lo = *(const v8bf*)p;
  v8bf hi = *(const v8bf*)(p + 16);
  v16bf b;
#pragma unroll
  for (int i = 0; i < 8; ++i) { b[i] = lo[i]; b[i + 8] = hi[i]; }
  return b;
}

// ------------------------------------------------- Tensor Data Mover (2D tile)
// Loads [rows][tile_x] bf16 tile (row stride = stride_elems) from global into
// contiguous LDS at lds_byte_off.  D# per CDNA5 ISA ch.8 (groups 0..3).
__device__ __forceinline__ uint32_t lds_off_of(const void* p) {
  // flat LDS pointer: aperture in [63:32], LDS byte offset in [31:0]
  return (uint32_t)(uintptr_t)p;
}

__device__ __forceinline__ void tdm_load_2d(uint32_t lds_byte_off,
                                            const __bf16* gptr,
                                            int tile_x, int rows,
                                            int stride_elems,
                                            int tensor_d0, int tensor_d1) {
  uint64_t ga = (uint64_t)(uintptr_t)gptr;
  v4u g0;
  g0[0] = 1u;                                   // count=1, user descriptor
  g0[1] = lds_byte_off;                         // lds_addr
  g0[2] = (unsigned)ga;                         // global_addr[31:0]
  g0[3] = (unsigned)(ga >> 32) | (2u << 30);    // global_addr[56:32] | type=2
  v8i g1;
  g1[0] = (1 << 16);                            // workgroup_mask=0, data_size=1 (2B)
  g1[1] = (tensor_d0 & 0xFFFF) << 16;           // barrier_addr=0 | tensor_dim0 lo
  g1[2] = ((tensor_d0 >> 16) & 0xFFFF) | ((tensor_d1 & 0xFFFF) << 16);
  g1[3] = ((tensor_d1 >> 16) & 0xFFFF) | (tile_x << 16);   // tile_dim0
  g1[4] = rows & 0xFFFF;                        // tile_dim1 | tile_dim2=0
  g1[5] = stride_elems;                         // tensor_dim0_stride[31:0]
  g1[6] = 0;                                    // stride[47:32] | dim1_stride lo
  g1[7] = 0;
  v4i g2 = {0, 0, 0, 0};
  v4i g3 = {0, 0, 0, 0};
  v8i gx = {0, 0, 0, 0, 0, 0, 0, 0};            // clang-23 6-arg form: extra group
  __builtin_amdgcn_tensor_load_to_lds(g0, g1, g2, g3, gx, 0);
}

// ---------------------------------------------------------------- prep kernels

__global__ void k_convert_scale(const float* __restrict__ src, __bf16* __restrict__ dst,
                                int n, float scale) {
  int i = blockIdx.x * blockDim.x + threadIdx.x;
  if (i < n) dst[i] = (__bf16)(src[i] * scale);
}

// w_depth (M,128,1,9) -> wdr[o][tap*128 + i], scaled by 1/sqrt(1152)
__global__ void k_reorder_wd(const float* __restrict__ wd, __bf16* __restrict__ dst) {
  int idx = blockIdx.x * blockDim.x + threadIdx.x;
  if (idx >= MCH * KDEP) return;
  int o  = idx / KDEP;
  int kk = idx - o * KDEP;
  int tap = kk >> 7;
  int i   = kk & 127;
  dst[idx] = (__bf16)(wd[((size_t)o * 128 + i) * 9 + tap] * INV_SQRT1152);
}

__global__ void k_cvec(const float* __restrict__ wemb, const float* __restrict__ emb,
                       const float* __restrict__ gain, float* __restrict__ cv) {
  int m = blockIdx.x * blockDim.x + threadIdx.x;
  if (m >= MCH) return;
  float s = 0.f;
  for (int e = 0; e < CCH; ++e) s += wemb[(size_t)m * CCH + e] * emb[e];
  cv[m] = 1.0f + s * gain[0] * INV_SQRT512;
}

__global__ void k_norm_x(const float* __restrict__ x, float* __restrict__ xn,
                         __bf16* __restrict__ xnb) {
  int l = blockIdx.x * blockDim.x + threadIdx.x;
  if (l >= LL) return;
  float s = 0.f;
  for (int c = 0; c < CCH; ++c) { float v = x[(size_t)c * LL + l]; s += v * v; }
  float inv = 1.0f / (EPSN + sqrtf(s) * INV_SQRT512);
  for (int c = 0; c < CCH; ++c) {
    float v = x[(size_t)c * LL + l] * inv;
    xn[(size_t)l * CCH + c]  = v;
    xnb[(size_t)l * CCH + c] = (__bf16)v;
  }
}

// ----------------------------------------------- TDM-staged GEMM core (NT form)
// Block = 8 waves sharing one 64-col B strip; wave w computes rows l0=bRow*128+w*16.
// B tile (64 x 128 bf16, 16KB) double-buffered in LDS via tensor_load_to_lds.

#define GEMM_PROLOG(Nn_)                                                     \
  __shared__ __attribute__((aligned(32))) __bf16 Bs[2][64 * 128];            \
  int lane = threadIdx.x & 31;                                               \
  int wave = threadIdx.x >> 5;                                               \
  int nTiles = (Nn_) >> 6;                                                   \
  int n0 = (blockIdx.x % nTiles) * 64;                                       \
  int l0 = (blockIdx.x / nTiles) * 128 + wave * 16;                          \
  int rbase = (lane < 16) ? 0 : 8;                                           \
  (void)wave;

#define GEMM_STAGE(Bptr_, K_, Nn_, kc_, NC_)                                 \
  if (threadIdx.x < 32) {                                                    \
    if ((kc_) + 1 < (NC_)) {                                                 \
      tdm_load_2d(lds_off_of(&Bs[((kc_) + 1) & 1][0]),                       \
                  (Bptr_) + (size_t)n0 * (K_) + ((kc_) + 1) * 128,           \
                  128, 64, (K_), (K_), (Nn_));                               \
      __builtin_amdgcn_s_wait_tensorcnt(1);                                  \
    } else {                                                                 \
      __builtin_amdgcn_s_wait_tensorcnt(0);                                  \
    }                                                                        \
  }                                                                          \
  __syncthreads();

// hoisted-fragment inner step: 1 A frag + 4 B frags, then 4 back-to-back WMMAs
#define GEMM_STEP(Ab_, lda_, Bl_, kbase_, kk_)                               \
  {                                                                          \
    v16bf a  = frag_a((Ab_), (lda_), lane, (kbase_) + (kk_));                \
    v16bf b0 = frag_b((Bl_) + (size_t)0 * 16 * 128, 128, lane, (kk_));       \
    v16bf b1 = frag_b((Bl_) + (size_t)1 * 16 * 128, 128, lane, (kk_));       \
    v16bf b2 = frag_b((Bl_) + (size_t)2 * 16 * 128, 128, lane, (kk_));       \
    v16bf b3 = frag_b((Bl_) + (size_t)3 * 16 * 128, 128, lane, (kk_));       \
    acc[0] = wmma_bf16(a, b0, acc[0]);                                       \
    acc[1] = wmma_bf16(a, b1, acc[1]);                                       \
    acc[2] = wmma_bf16(a, b2, acc[2]);                                       \
    acc[3] = wmma_bf16(a, b3, acc[3]);                                       \
  }

__global__ void __launch_bounds__(256) k_gemm_f32out(const __bf16* __restrict__ A,
                                                     const __bf16* __restrict__ B,
                                                     float* __restrict__ D,
                                                     int K, int Nn) {
  GEMM_PROLOG(Nn)
  const __bf16* Ab = A + (size_t)l0 * K;
  int NC = K >> 7;
  if (threadIdx.x < 32)
    tdm_load_2d(lds_off_of(&Bs[0][0]), B + (size_t)n0 * K, 128, 64, K, K, Nn);
  v8f acc[4] = {};
  for (int kc = 0; kc < NC; ++kc) {
    GEMM_STAGE(B, K, Nn, kc, NC)
    const __bf16* Bl = &Bs[kc & 1][0];
#pragma unroll
    for (int kk = 0; kk < 128; kk += 32)
      GEMM_STEP(Ab, K, Bl, kc * 128, kk)
    __syncthreads();
  }
#pragma unroll
  for (int t = 0; t < 4; ++t)
#pragma unroll
    for (int r = 0; r < 8; ++r)
      D[(size_t)(l0 + rbase + r) * Nn + n0 + t * 16 + (lane & 15)] = acc[t][r];
}

__global__ void __launch_bounds__(256) k_gemm_bf16out(const __bf16* __restrict__ A,
                                                      const __bf16* __restrict__ B,
                                                      __bf16* __restrict__ D,
                                                      int K, int Nn) {
  GEMM_PROLOG(Nn)
  const __bf16* Ab = A + (size_t)l0 * K;
  int NC = K >> 7;
  if (threadIdx.x < 32)
    tdm_load_2d(lds_off_of(&Bs[0][0]), B + (size_t)n0 * K, 128, 64, K, K, Nn);
  v8f acc[4] = {};
  for (int kc = 0; kc < NC; ++kc) {
    GEMM_STAGE(B, K, Nn, kc, NC)
    const __bf16* Bl = &Bs[kc & 1][0];
#pragma unroll
    for (int kk = 0; kk < 128; kk += 32)
      GEMM_STEP(Ab, K, Bl, kc * 128, kk)
    __syncthreads();
  }
#pragma unroll
  for (int t = 0; t < 4; ++t)
#pragma unroll
    for (int r = 0; r < 8; ++r)
      D[(size_t)(l0 + rbase + r) * Nn + n0 + t * 16 + (lane & 15)] = (__bf16)acc[t][r];
}

// grouped 1x9 conv (groups=8) as GEMM, K = 9 taps x 128 ch; fused c-mod + mp_silu
__global__ void __launch_bounds__(256) k_depth(const __bf16* __restrict__ y1t,
                                               const __bf16* __restrict__ wdr,
                                               const float* __restrict__ cv,
                                               __bf16* __restrict__ y2t) {
  GEMM_PROLOG(MCH)
  int g = n0 >> 7;
  int lrow = l0 + (lane & 15);
  int wcol = lrow & 511;
  int koA  = (lane < 16) ? 0 : 8;
  if (threadIdx.x < 32)
    tdm_load_2d(lds_off_of(&Bs[0][0]), wdr + (size_t)n0 * KDEP, 128, 64, KDEP, KDEP, MCH);
  v8f acc[4] = {};
  for (int kc = 0; kc < 9; ++kc) {           // one 128-chunk per tap
    GEMM_STAGE(wdr, KDEP, MCH, kc, 9)
    int shift = kc - 4;
    bool valid = (unsigned)(wcol + shift) < 512u;
    const __bf16* Bl = &Bs[kc & 1][0];
    const __bf16* ap = y1t + (size_t)(lrow + shift) * MCH + g * 128 + koA;
#pragma unroll
    for (int kk = 0; kk < 128; kk += 32) {
      v8bf lo = {}, hi = {};
      if (valid) { lo = *(const v8bf*)(ap + kk); hi = *(const v8bf*)(ap + kk + 16); }
      v16bf a;
#pragma unroll
      for (int i = 0; i < 8; ++i) { a[i] = lo[i]; a[i + 8] = hi[i]; }
      v16bf b0 = frag_b(Bl + (size_t)0 * 16 * 128, 128, lane, kk);
      v16bf b1 = frag_b(Bl + (size_t)1 * 16 * 128, 128, lane, kk);
      v16bf b2 = frag_b(Bl + (size_t)2 * 16 * 128, 128, lane, kk);
      v16bf b3 = frag_b(Bl + (size_t)3 * 16 * 128, 128, lane, kk);
      acc[0] = wmma_bf16(a, b0, acc[0]);
      acc[1] = wmma_bf16(a, b1, acc[1]);
      acc[2] = wmma_bf16(a, b2, acc[2]);
      acc[3] = wmma_bf16(a, b3, acc[3]);
    }
    __syncthreads();
  }
#pragma unroll
  for (int t = 0; t < 4; ++t)
#pragma unroll
    for (int r = 0; r < 8; ++r) {
      int o = n0 + t * 16 + (lane & 15);
      float vm = acc[t][r] * cv[o];
      float y  = vm / (1.0f + __expf(-vm)) * INV_SILU;
      y2t[(size_t)(l0 + rbase + r) * MCH + o] = (__bf16)y;
    }
}

// res1 GEMM + residual + pe interleave epilogue
__global__ void __launch_bounds__(256) k_res1(const __bf16* __restrict__ y2t,
                                              const __bf16* __restrict__ w1b,
                                              const float* __restrict__ xn,
                                              const float* __restrict__ pos,
                                              float* __restrict__ xr,
                                              __bf16* __restrict__ xrb,
                                              __bf16* __restrict__ pet) {
  GEMM_PROLOG(CCH)
  const __bf16* Ab = y2t + (size_t)l0 * MCH;
  if (threadIdx.x < 32)
    tdm_load_2d(lds_off_of(&Bs[0][0]), w1b + (size_t)n0 * MCH, 128, 64, MCH, MCH, CCH);
  v8f acc[4] = {};
  for (int kc = 0; kc < 8; ++kc) {
    GEMM_STAGE(w1b, MCH, CCH, kc, 8)
    const __bf16* Bl = &Bs[kc & 1][0];
#pragma unroll
    for (int kk = 0; kk < 128; kk += 32)
      GEMM_STEP(Ab, MCH, Bl, kc * 128, kk)
    __syncthreads();
  }
#pragma unroll
  for (int t = 0; t < 4; ++t)
#pragma unroll
    for (int r = 0; r < 8; ++r) {
      int l = l0 + rbase + r;
      int c = n0 + t * 16 + (lane & 15);
      float v = (xn[(size_t)l * CCH + c] + acc[t][r]) * INV_SQRT2;
      xr[(size_t)l * CCH + c]  = v;
      xrb[(size_t)l * CCH + c] = (__bf16)v;
      pet[(size_t)l * MCH + 2 * c]     = (__bf16)v;
      pet[(size_t)l * MCH + 2 * c + 1] = (__bf16)(v * pos[(size_t)c * LL + l]);
    }
}

// ------------------------------------------------------------- normalizations
__global__ void k_qk_norm(const float* __restrict__ qk, __bf16* __restrict__ qn,
                          __bf16* __restrict__ kn) {
  int idx = blockIdx.x * blockDim.x + threadIdx.x;
  if (idx >= LL * 8) return;
  int l = idx >> 3, rem = idx & 7, h = rem >> 1, sel = rem & 1;
  const float* row = qk + (size_t)l * MCH + h * 256 + sel;
  float s = 0.f;
  for (int d = 0; d < 128; ++d) { float v = row[2 * d]; s += v * v; }
  float inv = 1.0f / (EPSN + sqrtf(s) * INV_SQRT128);
  __bf16* dst = (sel ? kn : qn) + ((size_t)h * LL + l) * 128;
  for (int d = 0; d < 128; ++d) dst[d] = (__bf16)(row[2 * d] * inv);
}

__global__ void k_v_norm(const float* __restrict__ vr, __bf16* __restrict__ vt) {
  int idx = blockIdx.x * blockDim.x + threadIdx.x;
  if (idx >= LL * 4) return;
  int l = idx >> 2, h = idx & 3;
  const float* row = vr + (size_t)l * CCH + h * 128;
  float s = 0.f;
  for (int d = 0; d < 128; ++d) { float v = row[d]; s += v * v; }
  float inv = 1.0f / (EPSN + sqrtf(s) * INV_SQRT128);
  for (int d = 0; d < 128; ++d)
    vt[((size_t)h * 128 + d) * LL + l] = (__bf16)(row[d] * inv);  // [h][d][L]
}

// ---------------------------------------------------- flash attention + output
__global__ void __launch_bounds__(128) k_flash(const __bf16* __restrict__ qn,
                                               const __bf16* __restrict__ kn,
                                               const __bf16* __restrict__ vt,
                                               const float* __restrict__ xr,
                                               float* __restrict__ out) {
  __shared__ __attribute__((aligned(32))) float  sS[4][16 * 32];
  __shared__ __attribute__((aligned(32))) __bf16 sP[4][16 * 32];
  __shared__ __attribute__((aligned(32))) float  sR[4][16];
  int w    = threadIdx.x >> 5;
  int lane = threadIdx.x & 31;
  int wid  = blockIdx.x * 4 + w;
  int h    = wid >> 9;
  int i0   = (wid & 511) * 16;

  v16bf qA[4];
#pragma unroll
  for (int dc = 0; dc < 4; ++dc)
    qA[dc] = frag_a(qn + ((size_t)h * LL + i0) * 128, 128, lane, dc * 32);

  v8f oacc[8] = {};
  float m_i = -3.0e38f, l_i = 0.f;
  int colb  = lane & 15;
  int rbase = (lane < 16) ? 0 : 8;
  int ri = lane & 15, ch = lane >> 4;

  const __bf16* Kb = kn + (size_t)h * LL * 128;
  for (int j0 = 0; j0 < LL; j0 += 32) {
    v8f s0 = {}, s1 = {};
#pragma unroll
    for (int dc = 0; dc < 4; ++dc) {
      v16bf b0 = frag_b(Kb + (size_t)j0 * 128, 128, lane, dc * 32);
      v16bf b1 = frag_b(Kb + (size_t)(j0 + 16) * 128, 128, lane, dc * 32);
      s0 = wmma_bf16(qA[dc], b0, s0);
      s1 = wmma_bf16(qA[dc], b1, s1);
    }
#pragma unroll
    for (int r = 0; r < 8; ++r) {
      sS[w][(rbase + r) * 32 + colb]      = s0[r] * INV_SQRT128;
      sS[w][(rbase + r) * 32 + 16 + colb] = s1[r] * INV_SQRT128;
    }
    __syncthreads();
    float vals[16], pmax = -3.0e38f;
#pragma unroll
    for (int cc = 0; cc < 16; ++cc) {
      vals[cc] = sS[w][ri * 32 + ch * 16 + cc];
      pmax = fmaxf(pmax, vals[cc]);
    }
    float rowmax = fmaxf(pmax, __shfl_xor(pmax, 16));
    float m_new  = fmaxf(m_i, rowmax);
    float corr   = __expf(m_i - m_new);
    float psum   = 0.f;
#pragma unroll
    for (int cc = 0; cc < 16; ++cc) {
      float p = __expf(vals[cc] - m_new);
      sP[w][ri * 32 + ch * 16 + cc] = (__bf16)p;
      psum += p;
    }
    l_i = l_i * corr + psum + __shfl_xor(psum, 16);
    m_i = m_new;
    if (lane < 16) sR[w][ri] = corr;
    __syncthreads();
    float cr[8];
#pragma unroll
    for (int r = 0; r < 8; ++r) cr[r] = sR[w][rbase + r];
#pragma unroll
    for (int t = 0; t < 8; ++t)
#pragma unroll
      for (int r = 0; r < 8; ++r) oacc[t][r] *= cr[r];
    int po = (lane < 16) ? 0 : 8;
    v8bf plo = *(const v8bf*)&sP[w][ri * 32 + po];
    v8bf phi = *(const v8bf*)&sP[w][ri * 32 + po + 16];
    v16bf pa;
#pragma unroll
    for (int i = 0; i < 8; ++i) { pa[i] = plo[i]; pa[i + 8] = phi[i]; }
    int ko = (lane < 16) ? 0 : 8;
#pragma unroll
    for (int t = 0; t < 8; ++t) {
      const __bf16* vp = vt + ((size_t)h * 128 + t * 16 + colb) * LL + j0 + ko;
      v8bf vlo = *(const v8bf*)vp;
      v8bf vhi = *(const v8bf*)(vp + 16);
      v16bf vb;
#pragma unroll
      for (int i = 0; i < 8; ++i) { vb[i] = vlo[i]; vb[i + 8] = vhi[i]; }
      oacc[t] = wmma_bf16(pa, vb, oacc[t]);
    }
    __syncthreads();
  }
  if (lane < 16) sR[w][ri] = l_i;
  __syncthreads();
  float lr[8];
#pragma unroll
  for (int r = 0; r < 8; ++r) lr[r] = 1.0f / sR[w][rbase + r];
#pragma unroll
  for (int t = 0; t < 8; ++t)
#pragma unroll
    for (int r = 0; r < 8; ++r) {
      int c = h * 128 + t * 16 + colb;
      int l = i0 + rbase + r;
      float v = (xr[(size_t)l * CCH + c] + oacc[t][r] * lr[r]) * INV_SQRT2;
      v = fminf(fmaxf(v, -CLIPV), CLIPV);
      out[(size_t)c * LL + l] = v;
    }
}

// --------------------------------------------------------------------- launch
extern "C" void kernel_launch(void* const* d_in, const int* in_sizes, int n_in,
                              void* d_out, int out_size, void* d_ws, size_t ws_size,
                              hipStream_t stream) {
  (void)in_sizes; (void)n_in; (void)out_size; (void)ws_size;
  const float* x      = (const float*)d_in[0];
  const float* emb    = (const float*)d_in[1];
  const float* pos    = (const float*)d_in[2];
  const float* egain  = (const float*)d_in[3];
  const float* wres0  = (const float*)d_in[4];
  const float* wdepth = (const float*)d_in[5];
  const float* wemb   = (const float*)d_in[6];
  const float* wres1  = (const float*)d_in[7];
  const float* wqk    = (const float*)d_in[8];
  const float* wv     = (const float*)d_in[9];
  float* out = (float*)d_out;

  unsigned char* ws = (unsigned char*)d_ws;
  size_t off = 0;
  auto alloc = [&](size_t bytes) -> void* {
    void* p = ws + off; off += (bytes + 255) & ~(size_t)255; return p;
  };
  float*  xn_f32 = (float*) alloc((size_t)LL * CCH * 4);
  __bf16* xn_bf  = (__bf16*)alloc((size_t)LL * CCH * 2);
  __bf16* y1t    = (__bf16*)alloc((size_t)LL * MCH * 2);   // aliased later: qkraw (32MB)
  __bf16* y2t    = (__bf16*)alloc((size_t)LL * MCH * 2);
  float*  xr_f32 = (float*) alloc((size_t)LL * CCH * 4);
  __bf16* xr_bf  = (__bf16*)alloc((size_t)LL * CCH * 2);
  __bf16* pet    = (__bf16*)alloc((size_t)LL * MCH * 2);   // aliased later: vraw (16MB)
  __bf16* qn     = (__bf16*)alloc((size_t)4 * LL * 128 * 2);
  __bf16* kn     = (__bf16*)alloc((size_t)4 * LL * 128 * 2);
  __bf16* vt     = (__bf16*)alloc((size_t)4 * 128 * LL * 2);
  __bf16* w0b    = (__bf16*)alloc((size_t)MCH * CCH * 2);
  __bf16* wdr    = (__bf16*)alloc((size_t)MCH * KDEP * 2);
  __bf16* w1b    = (__bf16*)alloc((size_t)CCH * MCH * 2);
  __bf16* wqkb   = (__bf16*)alloc((size_t)MCH * MCH * 2);
  __bf16* wvb    = (__bf16*)alloc((size_t)CCH * CCH * 2);
  float*  cv     = (float*) alloc((size_t)MCH * 4);
  float*  qkraw  = (float*)y1t;
  float*  vraw   = (float*)pet;

  // weight prep
  k_convert_scale<<<(MCH * CCH + 255) / 256, 256, 0, stream>>>(wres0, w0b, MCH * CCH, INV_SQRT512);
  k_convert_scale<<<(CCH * MCH + 255) / 256, 256, 0, stream>>>(wres1, w1b, CCH * MCH, INV_SQRT1024);
  k_convert_scale<<<(MCH * MCH + 255) / 256, 256, 0, stream>>>(wqk, wqkb, MCH * MCH, INV_SQRT1024);
  k_convert_scale<<<(CCH * CCH + 255) / 256, 256, 0, stream>>>(wv, wvb, CCH * CCH, INV_SQRT512);
  k_reorder_wd<<<(MCH * KDEP + 255) / 256, 256, 0, stream>>>(wdepth, wdr);
  k_cvec<<<(MCH + 255) / 256, 256, 0, stream>>>(wemb, emb, egain, cv);

  // pipeline  (grid = (L/128) * (N/64) blocks, 256 threads = 8 waves)
  k_norm_x<<<LL / 256, 256, 0, stream>>>(x, xn_f32, xn_bf);
  k_gemm_bf16out<<<64 * 16, 256, 0, stream>>>(xn_bf, w0b, y1t, CCH, MCH);       // res0
  k_depth<<<64 * 16, 256, 0, stream>>>(y1t, wdr, cv, y2t);                       // depth+silu
  k_res1<<<64 * 8, 256, 0, stream>>>(y2t, w1b, xn_f32, pos, xr_f32, xr_bf, pet); // res1+pe
  k_gemm_f32out<<<64 * 16, 256, 0, stream>>>(pet, wqkb, qkraw, MCH, MCH);        // qk
  k_qk_norm<<<(LL * 8) / 256, 256, 0, stream>>>(qkraw, qn, kn);
  k_gemm_f32out<<<64 * 8, 256, 0, stream>>>(xr_bf, wvb, vraw, CCH, CCH);         // v
  k_v_norm<<<(LL * 4) / 256, 256, 0, stream>>>(vraw, vt);
  k_flash<<<512, 128, 0, stream>>>(qn, kn, vt, xr_f32, out);                     // attention
}
